// PolicyNet_29944511987734
// MI455X (gfx1250) — matrix operand order repaired and checked by reading, not compiled, not run on hardware
//
#include <hip/hip_runtime.h>

typedef __attribute__((ext_vector_type(16))) _Float16 v16h;
typedef __attribute__((ext_vector_type(8)))  float    v8f;
typedef __attribute__((ext_vector_type(4)))  unsigned int u32x4;
typedef __attribute__((ext_vector_type(8)))  int      i32x8;
typedef __attribute__((ext_vector_type(4)))  int      i32x4;

union AFrag {
    v16h h;
    unsigned int u[8];
};

static __device__ inline v8f splat8(float v) {
    v8f r = {v, v, v, v, v, v, v, v};
    return r;
}

// ---------------------------------------------------------------------------
// Kernel P: pack f32 weights [N][K] into WMMA B-fragment order (f16).
// dst element e = ((kt*NT + nt)*32 + lane)*16 + s  holds  src[n*K + k]
//   n = nt*16 + (lane&15),  k = kt*32 + (lane>>4)*8 + (s&7) + (s>>3)*16
// (16-bit B 32x16 layout: lanes 0-15 hold K=0..7,16..23, lanes 16-31 hold K=8..15,24..31)
// ---------------------------------------------------------------------------
__global__ __launch_bounds__(256) void pack_wmma_b(const float* __restrict__ src,
                                                   _Float16* __restrict__ dst,
                                                   int K, int NT, int total) {
    int e = blockIdx.x * 256 + threadIdx.x;
    if (e >= total) return;
    int s    = e & 15;
    int lane = (e >> 4) & 31;
    int nt   = (e >> 9) % NT;
    int kt   = e / (512 * NT);
    int n = nt * 16 + (lane & 15);
    int k = kt * 32 + (lane >> 4) * 8 + (s & 7) + (s >> 3) * 16;
    dst[e] = (_Float16)src[n * K + k];
}

// ---------------------------------------------------------------------------
// Kernel 1: per-cell conv1 + conv2 (implicit GEMM via WMMA) -> 288 features.
// One block (8 waves) per cell. 8192 blocks.
//   conv1: M=100 (pad 112, 7 tiles), N=16, K=192 (6 steps)
//   conv2: M=9   (pad 16, 1 tile),  N=32 (2 tiles), K=256 (8 steps)
// ---------------------------------------------------------------------------
__global__ __launch_bounds__(256) void cell_cnn(const float* __restrict__ x,
                                                const float* __restrict__ b1,
                                                const float* __restrict__ b2,
                                                const _Float16* __restrict__ w1p,
                                                const _Float16* __restrict__ w2p,
                                                _Float16* __restrict__ feat) {
    __shared__ _Float16 ldsIn[3 * 44 * 44];   // padded cell input (f16)
    __shared__ _Float16 ldsC1[16 * 12 * 12];  // conv1 relu output, pad 1 (f16)

    const int tid  = threadIdx.x;
    const int cell = blockIdx.x;
    const int bimg = cell >> 6, p = cell & 63, pr = p >> 3, pc = p & 7;
    const float* cellBase = x + (size_t)bimg * 3 * 320 * 320 + (size_t)pr * 40 * 320 + (size_t)pc * 40;

    // Phase 0: stage padded input as f16; zero conv1-out ring.
    for (int i = tid; i < 3 * 44 * 44; i += 256) {
        int c = i / 1936, rem = i % 1936;
        int y = rem / 44 - 2, xx = rem % 44 - 2;
        float v = 0.f;
        if ((unsigned)y < 40u && (unsigned)xx < 40u)
            v = cellBase[(size_t)c * 320 * 320 + (size_t)y * 320 + xx];
        ldsIn[i] = (_Float16)v;
    }
    for (int i = tid; i < 16 * 12 * 12; i += 256) ldsC1[i] = (_Float16)0.f;
    __syncthreads();

    const int lane = tid & 31, wave = tid >> 5;
    const int lm = lane & 15, laneHi = lane >> 4;

    // Phase 1: conv1 (waves 0..6 each own one M-tile of 16 output positions)
    if (wave < 7) {
        int m0   = wave * 16 + lm;
        int pidx = m0 < 100 ? m0 : 0;
        int oy = pidx / 10, ox = pidx % 10;
        v8f acc = splat8(b1[lm]);  // bias pre-loaded (n = lane%16 = out channel)
        for (int kt = 0; kt < 6; ++kt) {
            AFrag a;
#pragma unroll
            for (int j = 0; j < 8; ++j) {  // dword j holds K = kb, kb+1
                int kb = (j < 4) ? (2 * j) : (16 + 2 * (j - 4));
                int k  = kt * 32 + kb + laneHi * 8;
                int ic = k >> 6, kr = k & 63, ky = kr >> 3, kx = kr & 7;
                int off = ic * 1936 + (oy * 4 + ky) * 44 + (ox * 4 + kx);  // even
                a.u[j] = *reinterpret_cast<const unsigned int*>(&ldsIn[off]);
            }
            v16h bfrag = *reinterpret_cast<const v16h*>(&w1p[(size_t)(kt * 32 + lane) * 16]);
            acc = __builtin_amdgcn_wmma_f32_16x16x32_f16(false, a.h, false, bfrag,
                                                         (short)0, acc, false, false);
        }
#pragma unroll
        for (int g = 0; g < 8; ++g) {
            int m = wave * 16 + g + laneHi * 8;
            if (m < 100) {
                int oy2 = m / 10, ox2 = m % 10;
                float v = acc[g];
                v = v > 0.f ? v : 0.f;  // relu
                ldsC1[lm * 144 + (oy2 + 1) * 12 + (ox2 + 1)] = (_Float16)v;
            }
        }
    }
    __syncthreads();

    // Phase 2: conv2 (waves 0..1 own N-tiles nt=0,1)
    if (wave < 2) {
        int nt = wave;
        int oc = nt * 16 + lm;
        int pidx = lm < 9 ? lm : 0;
        int oy = pidx / 3, ox = pidx % 3;
        v8f acc = splat8(b2[oc]);
        for (int kt = 0; kt < 8; ++kt) {
            AFrag a;
#pragma unroll
            for (int j = 0; j < 8; ++j) {
                int kb = (j < 4) ? (2 * j) : (16 + 2 * (j - 4));
                int k  = kt * 32 + kb + laneHi * 8;
                int ic = k >> 4, kr = k & 15, ky = kr >> 2, kx = kr & 3;
                int off = ic * 144 + (oy * 4 + ky) * 12 + (ox * 4 + kx);  // even
                a.u[j] = *reinterpret_cast<const unsigned int*>(&ldsC1[off]);
            }
            v16h bfrag = *reinterpret_cast<const v16h*>(&w2p[(size_t)((kt * 2 + nt) * 32 + lane) * 16]);
            acc = __builtin_amdgcn_wmma_f32_16x16x32_f16(false, a.h, false, bfrag,
                                                         (short)0, acc, false, false);
        }
        _Float16* fout = feat + (size_t)cell * 288;
#pragma unroll
        for (int g = 0; g < 8; ++g) {
            int m = g + laneHi * 8;
            if (m < 9) {
                float v = acc[g];
                v = v > 0.f ? v : 0.f;           // relu
                fout[oc * 9 + m] = (_Float16)v;  // NCHW flatten: ch*9 + pos
            }
        }
    }
}

// ---------------------------------------------------------------------------
// Kernel 2: fc1 (WMMA, M=32 cells/block, N=256, K=288) + fc2 + softmax.
// Feature block (32*288 f16 = 18 KB, contiguous) staged into LDS via the
// Tensor Data Mover (TENSOR_LOAD_TO_LDS, tracked with TENSORcnt).
// 256 blocks x 8 waves. Wave w: M-tile = w>>2, N-tiles = (w&3)*4 .. +3.
// ---------------------------------------------------------------------------
__global__ __launch_bounds__(256) void fc_probs(const _Float16* __restrict__ feat,
                                                const _Float16* __restrict__ f1p,
                                                const float* __restrict__ fb1,
                                                const float* __restrict__ w2,
                                                const float* __restrict__ fb2,
                                                float* __restrict__ probs) {
    __shared__ _Float16 ldsA[32 * 288];  // staged features (TDM destination)
    __shared__ _Float16 ldsF[32 * 264];  // relu(fc1) per cell, padded stride

    const int tid = threadIdx.x;
    const int cell0 = blockIdx.x * 32;

    // --- TDM: DMA 32 cells of features (1 row x 9216 f16 elements) to LDS ---
    if (tid == 0) {
        unsigned long long ga = (unsigned long long)(const void*)(feat + (size_t)cell0 * 288);
        unsigned la = (unsigned)(size_t)(void*)ldsA;  // flat LDS addr[31:0] = LDS offset
        u32x4 g0;
        g0[0] = 1u;                                   // count=1, no gather, user D#
        g0[1] = la;                                   // lds_addr (bytes)
        g0[2] = (unsigned)ga;                         // global_addr[31:0]
        g0[3] = ((unsigned)(ga >> 32) & 0x01FFFFFFu)  // global_addr[56:32]
                | (2u << 30);                         // type = 2 ("image")
        i32x8 g1;
        g1[0] = 0x00010000;                           // wg_mask=0, data_size=1 (2B)
        g1[1] = (int)(9216u << 16);                   // tensor_dim0 = 9216 elems
        g1[2] = 0x00010000;                           // tensor_dim1 = 1
        g1[3] = (int)(9216u << 16);                   // tile_dim0 = 9216 elems
        g1[4] = 1;                                    // tile_dim1 = 1
        g1[5] = 9216;                                 // tensor_dim0_stride
        g1[6] = 0;
        g1[7] = 0;
        i32x4 g2 = {0, 0, 0, 0};                      // <=2D tensor: groups 2/3 unused
        i32x4 g3 = {0, 0, 0, 0};
        i32x8 g4 = {0, 0, 0, 0, 0, 0, 0, 0};          // extended descriptor: unused
        __builtin_amdgcn_tensor_load_to_lds(g0, g1, g2, g3, g4, 0);
    }
    __builtin_amdgcn_s_wait_tensorcnt((short)0);
    __syncthreads();

    const int lane = tid & 31, wave = tid >> 5;
    const int lm = lane & 15, laneHi = lane >> 4;
    const int mt = wave >> 2;
    const int ntBase = (wave & 3) * 4;
    const int row = mt * 16 + lm;

    v8f acc[4];
#pragma unroll
    for (int i = 0; i < 4; ++i) acc[i] = splat8(fb1[(ntBase + i) * 16 + lm]);

    for (int kt = 0; kt < 9; ++kt) {
        AFrag a;
#pragma unroll
        for (int j = 0; j < 8; ++j) {
            int kb = (j < 4) ? (2 * j) : (16 + 2 * (j - 4));
            int k  = kt * 32 + kb + laneHi * 8;
            a.u[j] = *reinterpret_cast<const unsigned int*>(&ldsA[row * 288 + k]);
        }
        if (kt < 8)  // prefetch next K-step of B tiles
            __builtin_prefetch(&f1p[(size_t)(((kt + 1) * 16 + ntBase) * 32 + lane) * 16], 0, 0);
#pragma unroll
        for (int i = 0; i < 4; ++i) {
            int nt = ntBase + i;
            v16h bfrag = *reinterpret_cast<const v16h*>(&f1p[(size_t)((kt * 16 + nt) * 32 + lane) * 16]);
            acc[i] = __builtin_amdgcn_wmma_f32_16x16x32_f16(false, a.h, false, bfrag,
                                                            (short)0, acc[i], false, false);
        }
    }
#pragma unroll
    for (int i = 0; i < 4; ++i) {
        int n = (ntBase + i) * 16 + lm;
#pragma unroll
        for (int g = 0; g < 8; ++g) {
            int m = g + laneHi * 8;
            float v = acc[i][g];
            v = v > 0.f ? v : 0.f;
            ldsF[(mt * 16 + m) * 264 + n] = (_Float16)v;
        }
    }
    __syncthreads();

    // fc2 + softmax: one lane per cell
    if (tid < 32) {
        float o[4];
#pragma unroll
        for (int a = 0; a < 4; ++a) {
            float s = fb2[a];
            for (int k = 0; k < 256; ++k)
                s += (float)ldsF[tid * 264 + k] * w2[a * 256 + k];
            o[a] = s;
        }
        float mx = o[0];
#pragma unroll
        for (int a = 1; a < 4; ++a) mx = o[a] > mx ? o[a] : mx;
        float e[4], sum = 0.f;
#pragma unroll
        for (int a = 0; a < 4; ++a) { e[a] = __expf(o[a] - mx); sum += e[a]; }
        float inv = 1.f / sum;
        float* pout = probs + (size_t)(cell0 + tid) * 4;
#pragma unroll
        for (int a = 0; a < 4; ++a) pout[a] = e[a] * inv;
    }
}

// ---------------------------------------------------------------------------
// Kernel 3: per-image graph reasoning (max-product closure on 8x8 grid).
// 128 blocks x 256 threads.
// ---------------------------------------------------------------------------
__global__ __launch_bounds__(256) void reason(const float* __restrict__ probs,
                                              float* __restrict__ out) {
    __shared__ float aP[64], gP[64], eP[64];
    __shared__ float mA[64], mG[64], mE[64];
    __shared__ float nodeS[64], s1S[64], reachS[64];
    __shared__ float flagG[64], flagE[64];
    __shared__ float Mp[4096], Mq[4096];
    __shared__ float naS[4];

    const int tid = threadIdx.x;
    const int b   = blockIdx.x;
    const float* pb = probs + (size_t)b * 256;

    if (tid < 64) {
        aP[tid] = pb[tid * 4 + 0];
        gP[tid] = pb[tid * 4 + 1];
        eP[tid] = pb[tid * 4 + 2];
    }
    __syncthreads();

    // top-k retain masks (keep p iff #(values strictly greater) < k)
    if (tid < 192) {
        int which = tid >> 6, i = tid & 63;
        const float* arr = (which == 0) ? aP : ((which == 1) ? gP : eP);
        int k = (which == 2) ? 7 : 3;
        float v = arr[i];
        int cnt = 0;
        for (int j = 0; j < 64; ++j) cnt += (arr[j] > v) ? 1 : 0;
        float mv = (cnt < k) ? v : 0.f;
        if (which == 0) mA[i] = mv; else if (which == 1) mG[i] = mv; else mE[i] = mv;
    }
    __syncthreads();

    if (tid < 64) {
        float nd = 0.9f * (1.f - mE[tid]);
        nodeS[tid] = nd;
        s1S[tid]   = mA[tid] * nd;
    }
    __syncthreads();

    // top-2 goal / top-6 enemy selection flags (rank with index tiebreak)
    if (tid < 128) {
        int which = tid >> 6, j = tid & 63;
        const float* arr = which ? mE : mG;
        int k = which ? 6 : 2;
        float v = arr[j];
        int cnt = 0;
        for (int l = 0; l < 64; ++l) {
            float w = arr[l];
            cnt += (w > v || (w == v && l < j)) ? 1 : 0;
        }
        float f = (cnt < k) ? v : 1.0f;  // 1.0 = identity for product
        if (which) flagE[j] = f; else flagG[j] = f;
    }
    // init M[i][j] = (i==j) ? 1 : (adj ? node[j] : 0)
    for (int t = tid; t < 4096; t += 256) {
        int i = t >> 6, j = t & 63;
        float v;
        if (i == j) v = 1.f;
        else {
            int ri = i >> 3, ci = i & 7, rj = j >> 3, cj = j & 7;
            int dr = ri - rj; dr = dr < 0 ? -dr : dr;
            int dc = ci - cj; dc = dc < 0 ? -dc : dc;
            v = (dr + dc == 1) ? nodeS[j] : 0.f;
        }
        Mp[t] = v;
    }
    __syncthreads();

    // 6 max-product squarings, LDS ping-pong
    float* P = Mp;
    float* Q = Mq;
    const int ci = tid >> 2;
    const int j0 = (tid & 3) * 16;
    for (int rep = 0; rep < 6; ++rep) {
        float accv[16];
#pragma unroll
        for (int jj = 0; jj < 16; ++jj) accv[jj] = 0.f;
        for (int k = 0; k < 64; ++k) {
            float pik = P[ci * 64 + k];
#pragma unroll
            for (int jj = 0; jj < 16; ++jj) {
                float t2 = pik * P[k * 64 + j0 + jj];
                accv[jj] = accv[jj] > t2 ? accv[jj] : t2;
            }
        }
#pragma unroll
        for (int jj = 0; jj < 16; ++jj) Q[ci * 64 + j0 + jj] = accv[jj];
        __syncthreads();
        float* tswap = P; P = Q; Q = tswap;
    }

    // reach[p] = max_j M[p][j] * goal[j]
    if (tid < 64) {
        float r = 0.f;
        for (int j = 0; j < 64; ++j) {
            float t2 = P[tid * 64 + j] * mG[j];
            r = r > t2 ? r : t2;
        }
        reachS[tid] = r;
    }
    __syncthreads();

    // na[a] = max_j s1[prev(a,j)] * node[j] * reach[j]
    if (tid < 4) {
        int a = tid;
        float best = 0.f;
        for (int j = 0; j < 64; ++j) {
            int rj = j >> 3, cj = j & 7;
            int p1 = -1;
            if (a == 0 && rj <= 6) p1 = j + 8;       // up:    p2 = p1 - 8
            else if (a == 1 && cj >= 1) p1 = j - 1;  // right: p2 = p1 + 1
            else if (a == 2 && rj >= 1) p1 = j - 8;  // down:  p2 = p1 + 8
            else if (a == 3 && cj <= 6) p1 = j + 1;  // left:  p2 = p1 - 1
            if (p1 >= 0) {
                float t2 = s1S[p1] * nodeS[j] * reachS[j];
                best = best > t2 ? best : t2;
            }
        }
        naS[a] = best;
    }
    __syncthreads();

    if (tid == 0) {
        float mx = naS[0];
#pragma unroll
        for (int a = 1; a < 4; ++a) mx = naS[a] > mx ? naS[a] : mx;
        float e[4], sum = 0.f;
#pragma unroll
        for (int a = 0; a < 4; ++a) { e[a] = __expf(naS[a] - mx); sum += e[a]; }
        float inv = 1.f / sum;
#pragma unroll
        for (int a = 0; a < 4; ++a) out[b * 4 + a] = e[a] * inv;

        float tg = 1.f, te = 1.f;
        for (int j = 0; j < 64; ++j) { tg *= flagG[j]; te *= flagE[j]; }
        float viol = (tg > te ? tg : te) * 0.2f;
        out[512 + b] = viol;
    }
}

// ---------------------------------------------------------------------------
// Launch
// ---------------------------------------------------------------------------
extern "C" void kernel_launch(void* const* d_in, const int* in_sizes, int n_in,
                              void* d_out, int out_size, void* d_ws, size_t ws_size,
                              hipStream_t stream) {
    const float* x        = (const float*)d_in[0];
    const float* conv1_w  = (const float*)d_in[1];
    const float* conv1_b  = (const float*)d_in[2];
    const float* conv2_w  = (const float*)d_in[3];
    const float* conv2_b  = (const float*)d_in[4];
    const float* fc1_w    = (const float*)d_in[5];
    const float* fc1_b    = (const float*)d_in[6];
    const float* fc2_w    = (const float*)d_in[7];
    const float* fc2_b    = (const float*)d_in[8];

    char* ws = (char*)d_ws;
    _Float16* w1p  = (_Float16*)(ws + 0);        //   6 KB (6*32*16 halves)
    _Float16* w2p  = (_Float16*)(ws + 8192);     //  16 KB (8*2*32*16)
    _Float16* f1p  = (_Float16*)(ws + 24576);    // 144 KB (9*16*32*16)
    _Float16* feat = (_Float16*)(ws + 172032);   // 4.5 MB (8192*288)
    float*    prb  = (float*)  (ws + 4890624);   // 128 KB (8192*4)

    pack_wmma_b<<<(3072  + 255) / 256, 256, 0, stream>>>(conv1_w, w1p, 192, 1,  3072);
    pack_wmma_b<<<(8192  + 255) / 256, 256, 0, stream>>>(conv2_w, w2p, 256, 2,  8192);
    pack_wmma_b<<<(73728 + 255) / 256, 256, 0, stream>>>(fc1_w,   f1p, 288, 16, 73728);

    cell_cnn<<<8192, 256, 0, stream>>>(x, conv1_b, conv2_b, w1p, w2p, feat);
    fc_probs<<<256, 256, 0, stream>>>(feat, f1p, fc1_b, fc2_w, fc2_b, prb);
    reason<<<128, 256, 0, stream>>>(prb, (float*)d_out);
}
